// TransLayer_25383256720161
// MI455X (gfx1250) — matrix operand order repaired and checked by reading, not compiled, not run on hardware
//
#include <hip/hip_runtime.h>
#include <hip/hip_bf16.h>

// MI455X (gfx1250) agent-attention TransLayer.
// All matmuls via v_wmma_f32_16x16x32_bf16 (wave32). Flash-style online
// softmax for both attention passes so no [A x N] score matrix is ever
// materialized. qkv/attn intermediates kept bf16 (fits in 192MB L2).
// Tile staging uses GLOBAL_LOAD_ASYNC_TO_LDS (ASYNCcnt) when the builtin is
// available, falling back to vector register copies otherwise.

typedef __bf16 bf16;
typedef __bf16 bf16x16 __attribute__((ext_vector_type(16)));
typedef float  f32x8   __attribute__((ext_vector_type(8)));
typedef int    v4i     __attribute__((ext_vector_type(4)));

static __device__ __forceinline__ f32x8 wmma_bf16(bf16x16 a, bf16x16 b, f32x8 c) {
    return __builtin_amdgcn_wmma_f32_16x16x32_bf16(false, a, false, b, (short)0, c,
                                                   false, false);
}

// ---- async global->LDS staging (CDNA5 ASYNCcnt path) with safe fallback ----
#if __has_builtin(__builtin_amdgcn_global_load_async_to_lds_b128)
#define ASYNC_LDS 1
typedef __attribute__((address_space(1))) v4i* as1_v4i_p;  // global
typedef __attribute__((address_space(3))) v4i* as3_v4i_p;  // LDS
#endif

static __device__ __forceinline__ void cp_b128(void* lds, const void* g) {
#ifdef ASYNC_LDS
    __builtin_amdgcn_global_load_async_to_lds_b128((as1_v4i_p)g, (as3_v4i_p)lds,
                                                   0, 0);
#else
    *(uint4*)lds = *(const uint4*)g;
#endif
}

static __device__ __forceinline__ void cp_wait() {
#ifdef ASYNC_LDS
#if __has_builtin(__builtin_amdgcn_s_wait_asynccnt)
    __builtin_amdgcn_s_wait_asynccnt(0);
#else
    asm volatile("s_wait_asynccnt 0" ::: "memory");
#endif
#endif
}

// ---------------------------------------------------------------- converts
__global__ __launch_bounds__(256) void cvt_kernel(const float* __restrict__ in,
                                                  bf16* __restrict__ out, int n) {
    int i = blockIdx.x * 256 + threadIdx.x;
    if (i < n) out[i] = (bf16)in[i];
}

// ---------------------------------------------------------------- layernorm
__global__ __launch_bounds__(256) void ln_kernel(const float* __restrict__ x,
                                                 const float* __restrict__ g,
                                                 const float* __restrict__ b,
                                                 bf16* __restrict__ xn) {
    __shared__ float red[16];
    const int row  = blockIdx.x;
    const int tid  = threadIdx.x;
    const int wave = tid >> 5, lane = tid & 31;
    const float* xr = x + (long)row * 512;
    float v0 = xr[tid], v1 = xr[tid + 256];
    float s = v0 + v1, s2 = v0 * v0 + v1 * v1;
#pragma unroll
    for (int m = 1; m < 32; m <<= 1) {
        s  += __shfl_xor(s,  m, 32);
        s2 += __shfl_xor(s2, m, 32);
    }
    if (lane == 0) { red[wave] = s; red[8 + wave] = s2; }
    __syncthreads();
    if (tid == 0) {
        float ts = 0.f, ts2 = 0.f;
#pragma unroll
        for (int w = 0; w < 8; ++w) { ts += red[w]; ts2 += red[8 + w]; }
        red[0] = ts * (1.f / 512.f);
        red[1] = ts2 * (1.f / 512.f);
    }
    __syncthreads();
    const float mu   = red[0];
    const float rstd = rsqrtf(red[1] - mu * mu + 1e-5f);
    xn[(long)row * 512 + tid]       = (bf16)((v0 - mu) * rstd * g[tid]       + b[tid]);
    xn[(long)row * 512 + tid + 256] = (bf16)((v1 - mu) * rstd * g[tid + 256] + b[tid + 256]);
}

// ---------------------------------------------------------------- GEMM
// C[M,N] = A[M,K] * B[K,N], bf16 in, block tile 128x64, 8 waves, K-step 32.
#define GA_LD 40
#define GB_LD 72
template <bool FINAL>
__global__ __launch_bounds__(256) void gemm_kernel(
    const bf16* __restrict__ A, const bf16* __restrict__ B,
    bf16* __restrict__ Cb, float* __restrict__ Cf,
    const float* __restrict__ bias, const float* __restrict__ resid,
    int M, int N, int Kd, int lda, int ldb, int ldc) {
    __shared__ bf16 As[128 * GA_LD];
    __shared__ bf16 Bs[32 * GB_LD];
    const int tid  = threadIdx.x;
    const int wave = tid >> 5, lane = tid & 31;
    const int half = lane >> 4, lc = lane & 15;
    const int m0 = blockIdx.y * 128;
    const int n0 = blockIdx.x * 64;

    f32x8 acc[4];
#pragma unroll
    for (int t = 0; t < 4; ++t)
#pragma unroll
        for (int j = 0; j < 8; ++j) acc[t][j] = 0.f;

    for (int k0 = 0; k0 < Kd; k0 += 32) {
        __syncthreads();
        {   // stage A tile 128x32 (async -> LDS)
            int r   = tid >> 1;
            int off = (tid & 1) * 16;
            cp_b128(&As[r * GA_LD + off],     &A[(long)(m0 + r) * lda + k0 + off]);
            cp_b128(&As[r * GA_LD + off + 8], &A[(long)(m0 + r) * lda + k0 + off + 8]);
            if (k0 + 32 < Kd)
                __builtin_prefetch(&A[(long)(m0 + r) * lda + k0 + 32 + off], 0, 1);
            // stage B tile 32x64
            int rb = tid >> 3;
            int ob = (tid & 7) * 8;
            cp_b128(&Bs[rb * GB_LD + ob], &B[(long)(k0 + rb) * ldb + n0 + ob]);
        }
        cp_wait();
        __syncthreads();

        bf16x16 af;
        {
            int r  = wave * 16 + lc;
            int kb = half << 3;
#pragma unroll
            for (int i = 0; i < 8; ++i) af[i]     = As[r * GA_LD + kb + i];
#pragma unroll
            for (int i = 0; i < 8; ++i) af[8 + i] = As[r * GA_LD + kb + 16 + i];
        }
#pragma unroll
        for (int t = 0; t < 4; ++t) {
            bf16x16 bfr;
            int kb2 = half << 3;
#pragma unroll
            for (int i = 0; i < 16; ++i)
                bfr[i] = Bs[(kb2 + i + (i & 8)) * GB_LD + t * 16 + lc];
            acc[t] = wmma_bf16(af, bfr, acc[t]);
        }
    }

#pragma unroll
    for (int t = 0; t < 4; ++t)
#pragma unroll
        for (int i = 0; i < 8; ++i) {
            int r   = m0 + wave * 16 + i + 8 * half;
            int col = n0 + t * 16 + lc;
            float v = acc[t][i];
            if (FINAL) {
                Cf[(long)r * ldc + col] = v + bias[col] + resid[(long)r * ldc + col];
            } else {
                Cb[(long)r * ldc + col] = (bf16)v;
            }
        }
}

// ---------------------------------------------------------------- flash attn
// O[rows, 64] = softmax(Q @ K^T * scale) @ V  (+ optional residual R).
// Block = 128 threads = 4 waves; each wave owns 16 Q rows; streams keys in
// chunks of 64 with online (rescaled) softmax. d = 64 fixed.
#define FA_LD 72
__global__ __launch_bounds__(128) void fa_kernel(
    const bf16* __restrict__ Qb, const bf16* __restrict__ Kb,
    const bf16* __restrict__ Vb, const bf16* __restrict__ Rb,
    bf16* __restrict__ Ob,
    int sQ, int sK, int sV, int sR, int sO,
    long hsQ, long hsK, long hsV, long hsR, long hsO,
    int Nk, float scale) {
    __shared__ bf16 Ks[64 * FA_LD];
    __shared__ bf16 Vs[64 * FA_LD];
    __shared__ bf16 Ps[4][16 * FA_LD];

    const int h    = blockIdx.y;
    const int tid  = threadIdx.x;
    const int wave = tid >> 5, lane = tid & 31;
    const int half = lane >> 4, lc = lane & 15;
    const bf16* Q = Qb + h * hsQ;
    const bf16* K = Kb + h * hsK;
    const bf16* V = Vb + h * hsV;
    const bf16* R = Rb ? Rb + h * hsR : nullptr;
    bf16* O = Ob + h * hsO;
    const int row0 = blockIdx.x * 64 + wave * 16;

    // Q fragments (two 16x32 A-frags covering d=64), loaded once.
    bf16x16 qf[2];
#pragma unroll
    for (int s = 0; s < 2; ++s) {
        int r  = row0 + lc;
        int kb = s * 32 + (half << 3);
#pragma unroll
        for (int i = 0; i < 8; ++i) qf[s][i]     = Q[(long)r * sQ + kb + i];
#pragma unroll
        for (int i = 0; i < 8; ++i) qf[s][8 + i] = Q[(long)r * sQ + kb + 16 + i];
    }

    float m[8], l[8];
    f32x8 o[4];
#pragma unroll
    for (int i = 0; i < 8; ++i) { m[i] = -1e30f; l[i] = 0.f; }
#pragma unroll
    for (int t = 0; t < 4; ++t)
#pragma unroll
        for (int j = 0; j < 8; ++j) o[t][j] = 0.f;

    bf16* P = &Ps[wave][0];

    for (int c0 = 0; c0 < Nk; c0 += 64) {
        __syncthreads();
        {   // stage K/V chunk: 64 rows x 64 of d (async -> LDS)
            int r   = tid >> 1;
            int off = (tid & 1) * 32;
            const bf16* kg = &K[(long)(c0 + r) * sK + off];
            const bf16* vg = &V[(long)(c0 + r) * sV + off];
            bf16* kd = &Ks[r * FA_LD + off];
            bf16* vd = &Vs[r * FA_LD + off];
#pragma unroll
            for (int i = 0; i < 4; ++i) {
                cp_b128(kd + i * 8, kg + i * 8);
                cp_b128(vd + i * 8, vg + i * 8);
            }
            if (c0 + 64 < Nk) {
                __builtin_prefetch(&K[(long)(c0 + 64 + r) * sK + off], 0, 1);
                __builtin_prefetch(&V[(long)(c0 + 64 + r) * sV + off], 0, 1);
            }
        }
        cp_wait();
        __syncthreads();

        // S = Q @ K^T * scale : 4 tiles of 16x16 keys.
        f32x8 sacc[4];
#pragma unroll
        for (int t = 0; t < 4; ++t) {
            f32x8 z;
#pragma unroll
            for (int j = 0; j < 8; ++j) z[j] = 0.f;
#pragma unroll
            for (int s = 0; s < 2; ++s) {
                bf16x16 bfr;   // B[kd, keycol] = Ks[key][kd]
                int kb = s * 32 + (half << 3);
#pragma unroll
                for (int i = 0; i < 16; ++i)
                    bfr[i] = Ks[(t * 16 + lc) * FA_LD + kb + i + (i & 8)];
                z = wmma_bf16(qf[s], bfr, z);
            }
#pragma unroll
            for (int j = 0; j < 8; ++j) sacc[t][j] = z[j] * scale;
        }

        // online softmax: row max across 64 keys
        float alpha[8];
#pragma unroll
        for (int i = 0; i < 8; ++i) {
            float v = fmaxf(fmaxf(sacc[0][i], sacc[1][i]),
                            fmaxf(sacc[2][i], sacc[3][i]));
#pragma unroll
            for (int msk = 1; msk < 16; msk <<= 1)
                v = fmaxf(v, __shfl_xor(v, msk, 32));
            float mn = fmaxf(m[i], v);
            alpha[i] = __expf(m[i] - mn);
            m[i] = mn;
        }
        // p = exp(s - m), stash bf16 P tile for WMMA re-feed
#pragma unroll
        for (int t = 0; t < 4; ++t)
#pragma unroll
            for (int i = 0; i < 8; ++i) {
                float p = __expf(sacc[t][i] - m[i]);
                sacc[t][i] = p;
                P[(i + 8 * half) * FA_LD + t * 16 + lc] = (bf16)p;
            }
#pragma unroll
        for (int i = 0; i < 8; ++i) {
            float s = sacc[0][i] + sacc[1][i] + sacc[2][i] + sacc[3][i];
#pragma unroll
            for (int msk = 1; msk < 16; msk <<= 1)
                s += __shfl_xor(s, msk, 32);
            l[i] = l[i] * alpha[i] + s;
        }
#pragma unroll
        for (int t = 0; t < 4; ++t)
#pragma unroll
            for (int i = 0; i < 8; ++i) o[t][i] *= alpha[i];

        // wave-private LDS store->load of P: enforce DS ordering explicitly
        asm volatile("s_wait_dscnt 0" ::: "memory");

        // O += P @ V
#pragma unroll
        for (int s = 0; s < 2; ++s) {
            bf16x16 pa;
            {
                int kb = s * 32 + (half << 3);
#pragma unroll
                for (int i = 0; i < 8; ++i) pa[i]     = P[lc * FA_LD + kb + i];
#pragma unroll
                for (int i = 0; i < 8; ++i) pa[8 + i] = P[lc * FA_LD + kb + 16 + i];
            }
#pragma unroll
            for (int t = 0; t < 4; ++t) {
                bf16x16 vb;
                int kb2 = s * 32 + (half << 3);
#pragma unroll
                for (int i = 0; i < 16; ++i)
                    vb[i] = Vs[(kb2 + i + (i & 8)) * FA_LD + t * 16 + lc];
                o[t] = wmma_bf16(pa, vb, o[t]);
            }
        }
    }

    // epilogue: /l, + optional value residual, bf16 store
#pragma unroll
    for (int t = 0; t < 4; ++t)
#pragma unroll
        for (int i = 0; i < 8; ++i) {
            int r   = row0 + i + 8 * half;
            int col = t * 16 + lc;
            float val = o[t][i] / l[i];
            if (R) val += (float)R[(long)r * sR + col];
            O[(long)r * sO + col] = (bf16)val;
        }
}

// ---------------------------------------------------------------- launcher
extern "C" void kernel_launch(void* const* d_in, const int* in_sizes, int n_in,
                              void* d_out, int out_size, void* d_ws, size_t ws_size,
                              hipStream_t stream) {
    const float* x     = (const float*)d_in[0];
    const float* g     = (const float*)d_in[1];
    const float* be    = (const float*)d_in[2];
    const float* wqkv  = (const float*)d_in[3];
    const float* agent = (const float*)d_in[4];
    const float* wout  = (const float*)d_in[5];
    const float* bout  = (const float*)d_in[6];
    float* out = (float*)d_out;

    char* ws = (char*)d_ws;
    bf16* xn     = (bf16*)(ws);                 // 16384*512       = 16 MB
    bf16* wqkvb  = (bf16*)(ws + 16777216);      // 512*1536        = 1.5 MB
    bf16* agentb = (bf16*)(ws + 18350080);      // 8*512*64        = 0.5 MB
    bf16* woutb  = (bf16*)(ws + 18874368);      // 512*512         = 0.5 MB
    bf16* qkv    = (bf16*)(ws + 19398656);      // 16384*1536      = 48 MB
    bf16* agv    = (bf16*)(ws + 69730304);      // 8*512*64        = 0.5 MB
    bf16* attn   = (bf16*)(ws + 70254592);      // 16384*512       = 16 MB

    cvt_kernel<<<(786432 + 255) / 256, 256, 0, stream>>>(wqkv, wqkvb, 786432);
    cvt_kernel<<<(262144 + 255) / 256, 256, 0, stream>>>(agent, agentb, 262144);
    cvt_kernel<<<(262144 + 255) / 256, 256, 0, stream>>>(wout, woutb, 262144);

    ln_kernel<<<16384, 256, 0, stream>>>(x, g, be, xn);

    // qkv = xn @ w_qkv   [16384,512] x [512,1536]
    gemm_kernel<false><<<dim3(1536 / 64, 16384 / 128), 256, 0, stream>>>(
        xn, wqkvb, qkv, nullptr, nullptr, nullptr,
        16384, 1536, 512, 512, 1536, 1536);

    const float scale = 0.125f;  // DIM_HEAD^-0.5

    // pass 1: agent_v[h] = softmax_N(agent @ k^T) @ v   (A=512 rows per head)
    fa_kernel<<<dim3(8, 8), 128, 0, stream>>>(
        agentb, qkv + 512, qkv + 1024, nullptr, agv,
        64, 1536, 1536, 0, 64,
        (long)512 * 64, 64, 64, 0, (long)512 * 64,
        16384, scale);

    // pass 2: out[h] = softmax_A(q @ agent^T) @ agent_v + v   (N=16384 rows)
    fa_kernel<<<dim3(256, 8), 128, 0, stream>>>(
        qkv, agentb, agv, qkv + 1024, attn,
        1536, 64, 64, 1536, 512,
        64, (long)512 * 64, (long)512 * 64, 64, 64,
        512, scale);

    // y = x + attn @ w_out + b_out
    gemm_kernel<true><<<dim3(512 / 64, 16384 / 128), 256, 0, stream>>>(
        attn, woutb, nullptr, out, bout, x,
        16384, 512, 512, 512, 512, 512);
}
